// GraphTransformerEncoder_39127152067284
// MI455X (gfx1250) — matrix operand order repaired and checked by reading, not compiled, not run on hardware
//
#include <hip/hip_runtime.h>
#include <math.h>

// ---------------------------------------------------------------------------
// Problem constants (match the JAX reference)
// ---------------------------------------------------------------------------
#define NN    4096
#define EE    131072
#define F_IN  3
#define HID   256
#define OUTD  512
#define HEADS 8
#define DH    32
#define EPS_BN 1e-5f

typedef __attribute__((ext_vector_type(16))) _Float16 v16h;
typedef __attribute__((ext_vector_type(8)))  float    v8f;

static __device__ __forceinline__ _Float16 f2h(float f) { return (_Float16)f; }

// B-fragment loader for v_wmma_f32_16x16x32_f16 (32x16 B tile).
// Lane holds column n = lane&15; e=0..7 -> K = kb+hs*8+e ; e=8..15 -> K = kb+16+hs*8+(e-8).
template<bool TB>
static __device__ __forceinline__ v16h load_bfrag(const float* __restrict__ B,
                                                  int ldb, int n, int kb, int hs) {
    v16h b;
    if (TB) {
        // B stored [N,K] row-major: two runs of 8 contiguous floats -> 4x b128 loads
        const float* p = &B[(size_t)n * ldb + kb + hs * 8];
        float4 q0 = *(const float4*)(p + 0);
        float4 q1 = *(const float4*)(p + 4);
        float4 q2 = *(const float4*)(p + 16);
        float4 q3 = *(const float4*)(p + 20);
        b[0] = f2h(q0.x);  b[1] = f2h(q0.y);  b[2]  = f2h(q0.z);  b[3]  = f2h(q0.w);
        b[4] = f2h(q1.x);  b[5] = f2h(q1.y);  b[6]  = f2h(q1.z);  b[7]  = f2h(q1.w);
        b[8] = f2h(q2.x);  b[9] = f2h(q2.y);  b[10] = f2h(q2.z);  b[11] = f2h(q2.w);
        b[12] = f2h(q3.x); b[13] = f2h(q3.y); b[14] = f2h(q3.z);  b[15] = f2h(q3.w);
    } else {
        // B stored [K,N] row-major: column-strided scalar loads
#pragma unroll
        for (int e = 0; e < 16; ++e)
            b[e] = f2h(B[(size_t)(kb + ((e < 8) ? 0 : 16) + hs * 8 + (e & 7)) * ldb + n]);
    }
    return b;
}

// ---------------------------------------------------------------------------
// WMMA f16 GEMM:  C[M,N] (+bias[n]) = (alpha*A) @ B
//   TA: A stored [K,M] (use A^T), else [M,K];  TB: B stored [N,K] (use B^T)
// fp32 storage; converted to f16 in-register for v_wmma; fp32 accumulate.
// Block = (32 lanes, 4 waves). Each wave owns a 16x32 C tile:
// one A fragment feeds two v_wmma (two 16x16 N tiles) per K step.
// grid = (M/64, N/32). Requires M%64==0, N%32==0, K%32==0.
// ---------------------------------------------------------------------------
template<bool TA, bool TB>
__global__ void gemm_wmma_f16(const float* __restrict__ A,
                              const float* __restrict__ B,
                              float* __restrict__ C,
                              const float* __restrict__ bias,
                              int M, int N, int K,
                              int lda, int ldb, int ldc, float alpha)
{
    const int lane = threadIdx.x;      // 0..31
    const int wave = threadIdx.y;      // 0..3
    const int r16  = lane & 15;
    const int hs   = lane >> 4;        // lane-half select (0/1)
    const int m0   = (blockIdx.x * 4 + wave) * 16;
    const int n0   = blockIdx.y * 32;
    if (m0 >= M) return;               // wave-uniform: EXEC stays all-ones

    // Bias is per-column and identical across all 8 accumulator rows:
    // load once, add unconditionally in the epilogue.
    const float bias0 = bias ? bias[n0 + r16]      : 0.0f;
    const float bias1 = bias ? bias[n0 + 16 + r16] : 0.0f;

    v8f acc0 = {}, acc1 = {};
    for (int kb = 0; kb < K; kb += 32) {
        // Speculative next-slab prefetch (global_prefetch_b8; faults dropped).
        {
            const float* pa = TA ? &A[(size_t)(kb + 32) * lda + (m0 + r16)]
                                 : &A[(size_t)(m0 + r16) * lda + (kb + 32)];
            __builtin_prefetch(pa, 0, 1);
        }

        // ---- A fragment: lane row m = m0 + (lane&15);
        //      e=0..7 -> K = kb+hs*8+e ; e=8..15 -> K = kb+16+hs*8+(e-8)
        v16h af;
        if (TA) {
#pragma unroll
            for (int e = 0; e < 16; ++e) {
                const int ka = kb + ((e < 8) ? 0 : 16) + hs * 8 + (e & 7);
                af[e] = f2h(alpha * A[(size_t)ka * lda + (m0 + r16)]);
            }
        } else {
            const float* p = &A[(size_t)(m0 + r16) * lda + kb + hs * 8];
            float4 a0 = *(const float4*)(p + 0);
            float4 a1 = *(const float4*)(p + 4);
            float4 a2 = *(const float4*)(p + 16);
            float4 a3 = *(const float4*)(p + 20);
            af[0] = f2h(alpha * a0.x);  af[1] = f2h(alpha * a0.y);
            af[2] = f2h(alpha * a0.z);  af[3] = f2h(alpha * a0.w);
            af[4] = f2h(alpha * a1.x);  af[5] = f2h(alpha * a1.y);
            af[6] = f2h(alpha * a1.z);  af[7] = f2h(alpha * a1.w);
            af[8] = f2h(alpha * a2.x);  af[9] = f2h(alpha * a2.y);
            af[10] = f2h(alpha * a2.z); af[11] = f2h(alpha * a2.w);
            af[12] = f2h(alpha * a3.x); af[13] = f2h(alpha * a3.y);
            af[14] = f2h(alpha * a3.z); af[15] = f2h(alpha * a3.w);
        }

        // ---- two B fragments (N tiles n0 and n0+16), two WMMAs per K step
        v16h b0 = load_bfrag<TB>(B, ldb, n0 + r16, kb, hs);
        v16h b1 = load_bfrag<TB>(B, ldb, n0 + 16 + r16, kb, hs);
        acc0 = __builtin_amdgcn_wmma_f32_16x16x32_f16(
                   false, af, false, b0, (short)0, acc0, false, false);
        acc1 = __builtin_amdgcn_wmma_f32_16x16x32_f16(
                   false, af, false, b1, (short)0, acc1, false, false);
    }

    // C/D layout: VGPR v -> row m0 + hs*8 + v, col = tile_n + (lane&15)
#pragma unroll
    for (int v = 0; v < 8; ++v) {
        const int m = m0 + hs * 8 + v;
        C[(size_t)m * ldc + n0 + r16]      = acc0[v] + bias0;
        C[(size_t)m * ldc + n0 + 16 + r16] = acc1[v] + bias1;
    }
}

// ---------------------------------------------------------------------------
// Elementwise / reduction helpers
// ---------------------------------------------------------------------------
__global__ void k_fill(float* p, float v, int n) {
    int i = blockIdx.x * blockDim.x + threadIdx.x;
    if (i < n) p[i] = v;
}

// h[N,HID] = x[N,3] @ W1[3,HID]
__global__ void k_xW1(const float* __restrict__ x, const float* __restrict__ W1,
                      float* __restrict__ h) {
    int i = blockIdx.x * blockDim.x + threadIdx.x;
    if (i >= NN * HID) return;
    int n = i >> 8, c = i & (HID - 1);
    h[i] = x[n * 3 + 0] * W1[0 * HID + c]
         + x[n * 3 + 1] * W1[1 * HID + c]
         + x[n * 3 + 2] * W1[2 * HID + c];
}

__global__ void k_deg_edges(const int* __restrict__ ei, float* __restrict__ deg) {
    int e = blockIdx.x * blockDim.x + threadIdx.x;
    if (e < EE) atomicAdd(&deg[ei[EE + e]], 1.0f);   // row1 = dst
}

__global__ void k_rsqrt(const float* __restrict__ deg, float* __restrict__ dis) {
    int i = blockIdx.x * blockDim.x + threadIdx.x;
    if (i < NN) dis[i] = rsqrtf(deg[i]);
}

// agg init: self-loop term + bias:  agg[n,c] = h[n,c]*dis[n]^2 + b[c]
__global__ void k_agg_init(const float* __restrict__ h, const float* __restrict__ dis,
                           const float* __restrict__ b, float* __restrict__ agg) {
    int i = blockIdx.x * blockDim.x + threadIdx.x;
    if (i >= NN * HID) return;
    int n = i >> 8, c = i & (HID - 1);
    float d = dis[n];
    agg[i] = h[i] * d * d + b[c];
}

// scatter-add messages: one block per edge, one thread per channel
__global__ void k_scatter(const float* __restrict__ h, const float* __restrict__ dis,
                          const int* __restrict__ ei, float* __restrict__ agg) {
    int e = blockIdx.x;
    int c = threadIdx.x;
    int s = ei[e], d = ei[EE + e];
    float w = dis[s] * dis[d];
    atomicAdd(&agg[(size_t)d * HID + c], h[(size_t)s * HID + c] * w);
}

__global__ void k_gelu(const float* __restrict__ in, float* __restrict__ out, int n) {
    int i = blockIdx.x * blockDim.x + threadIdx.x;
    if (i >= n) return;
    float v = in[i];
    out[i] = 0.5f * v * (1.0f + erff(v * 0.70710678118654752f));
}

// per-column mean / rsqrt(var+eps) over N rows; one block per column
__global__ void k_colstats(const float* __restrict__ g, float* __restrict__ mu,
                           float* __restrict__ rinv) {
    __shared__ float rs[256], rq[256];
    int c = blockIdx.x, t = threadIdx.x;
    float s = 0.f, q = 0.f;
    for (int r = t; r < NN; r += 256) {
        float v = g[(size_t)r * HID + c];
        s += v; q += v * v;
    }
    rs[t] = s; rq[t] = q;
    __syncthreads();
    for (int o = 128; o > 0; o >>= 1) {
        if (t < o) { rs[t] += rs[t + o]; rq[t] += rq[t + o]; }
        __syncthreads();
    }
    if (t == 0) {
        float m = rs[0] * (1.0f / NN);
        float var = rq[0] * (1.0f / NN) - m * m;     // biased, matches torch BN
        mu[c] = m;
        rinv[c] = rsqrtf(var + EPS_BN);
    }
}

__global__ void k_bn(const float* __restrict__ g, const float* __restrict__ mu,
                     const float* __restrict__ rinv, const float* __restrict__ gamma,
                     const float* __restrict__ beta, float* __restrict__ out) {
    int i = blockIdx.x * blockDim.x + threadIdx.x;
    if (i >= NN * HID) return;
    int c = i & (HID - 1);
    out[i] = (g[i] - mu[c]) * rinv[c] * gamma[c] + beta[c];
}

// row softmax in-place on scores[N,N] and accumulate mean over heads:
// meanA[m,n] += softmax(scores[m,:])[n] / HEADS.  One block per row.
__global__ void k_softmax_meanacc(float* __restrict__ scores,
                                  float* __restrict__ meanA) {
    __shared__ float red[256];
    int m = blockIdx.x, t = threadIdx.x;
    float* row = scores + (size_t)m * NN;

    float mx = -3.4e38f;
    for (int n = t; n < NN; n += 256) mx = fmaxf(mx, row[n]);
    red[t] = mx; __syncthreads();
    for (int o = 128; o > 0; o >>= 1) {
        if (t < o) red[t] = fmaxf(red[t], red[t + o]);
        __syncthreads();
    }
    mx = red[0]; __syncthreads();

    float s = 0.f;
    for (int n = t; n < NN; n += 256) {
        float p = __expf(row[n] - mx);
        row[n] = p;
        s += p;
    }
    red[t] = s; __syncthreads();
    for (int o = 128; o > 0; o >>= 1) {
        if (t < o) red[t] += red[t + o];
        __syncthreads();
    }
    float inv = 1.0f / red[0];

    float* ma = meanA + (size_t)m * NN;
    for (int n = t; n < NN; n += 256) {
        float p = row[n] * inv;
        row[n] = p;                       // normalized A (f32, fed to WMMA A@V)
        ma[n] += p * (1.0f / HEADS);
    }
}

// ---------------------------------------------------------------------------
// Host-side orchestration
// ---------------------------------------------------------------------------
extern "C" void kernel_launch(void* const* d_in, const int* in_sizes, int n_in,
                              void* d_out, int out_size, void* d_ws, size_t ws_size,
                              hipStream_t stream) {
    (void)in_sizes; (void)n_in; (void)out_size; (void)ws_size;

    const float* x      = (const float*)d_in[0];
    const int*   ei     = (const int*)  d_in[1];
    const float* W1     = (const float*)d_in[2];
    const float* b1     = (const float*)d_in[3];
    const float* W2     = (const float*)d_in[4];
    const float* b2     = (const float*)d_in[5];
    const float* gamma  = (const float*)d_in[6];
    const float* beta   = (const float*)d_in[7];
    const float* inW    = (const float*)d_in[8];   // [768,256]
    const float* inB    = (const float*)d_in[9];
    const float* outW   = (const float*)d_in[10];  // [256,256]
    const float* outB   = (const float*)d_in[11];
    const float* projW  = (const float*)d_in[12];  // [512,256]
    const float* projB  = (const float*)d_in[13];
    float* out = (float*)d_out;                    // [4096,512] fp32

    // workspace layout (fp32 elements)
    const size_t NF = (size_t)NN * HID;            // 1,048,576
    float* ws     = (float*)d_ws;
    float* bufA   = ws;                 // [N,HID]
    float* bufB   = ws + 1 * NF;        // [N,HID]
    float* bufC   = ws + 2 * NF;        // [N,HID]
    float* qkv    = ws + 3 * NF;        // [N,768]
    float* attn   = ws + 6 * NF;        // [N,HID]
    float* attn2  = ws + 7 * NF;        // [N,HID]
    float* pooled = ws + 8 * NF;        // [N,HID]
    float* deg    = ws + 9 * NF;        // [N]
    float* dis    = deg + NN;           // [N]
    float* mu     = dis + NN;           // [HID]
    float* rinv   = mu + HID;           // [HID]
    float* scores = ws + 10 * NF;       // [N,N]  (64 MB, L2-resident)
    float* meanA  = ws + 26 * NF;       // [N,N]

    const dim3 gb(32, 4);                          // 4 waves/block
    const int  EW = 256;
    const int  NHB = (int)((NF + EW - 1) / EW);

    // ---- degree / normalization (shared by both GCN layers) ----
    k_fill<<<(NN + EW - 1) / EW, EW, 0, stream>>>(deg, 1.0f, NN);   // self-loop
    k_deg_edges<<<(EE + EW - 1) / EW, EW, 0, stream>>>(ei, deg);
    k_rsqrt<<<(NN + EW - 1) / EW, EW, 0, stream>>>(deg, dis);

    // ---- GCN layer 1 ----
    k_xW1<<<NHB, EW, 0, stream>>>(x, W1, bufA);
    k_agg_init<<<NHB, EW, 0, stream>>>(bufA, dis, b1, bufB);
    k_scatter<<<EE, HID, 0, stream>>>(bufA, dis, ei, bufB);
    k_gelu<<<NHB, EW, 0, stream>>>(bufB, bufA, (int)NF);
    k_colstats<<<HID, EW, 0, stream>>>(bufA, mu, rinv);
    k_bn<<<NHB, EW, 0, stream>>>(bufA, mu, rinv, gamma, beta, bufC);

    // ---- GCN layer 2 (WMMA GEMM for h @ W2) ----
    gemm_wmma_f16<false, false><<<dim3(NN / 64, HID / 32), gb, 0, stream>>>(
        bufC, W2, bufA, nullptr, NN, HID, HID, HID, HID, HID, 1.0f);
    k_agg_init<<<NHB, EW, 0, stream>>>(bufA, dis, b2, bufB);
    k_scatter<<<EE, HID, 0, stream>>>(bufA, dis, ei, bufB);
    k_gelu<<<NHB, EW, 0, stream>>>(bufB, bufA, (int)NF);
    k_colstats<<<HID, EW, 0, stream>>>(bufA, mu, rinv);
    k_bn<<<NHB, EW, 0, stream>>>(bufA, mu, rinv, gamma, beta, bufC);

    // ---- QKV projection: qkv[N,768] = h @ inW^T + inB ----
    gemm_wmma_f16<false, true><<<dim3(NN / 64, (3 * HID) / 32), gb, 0, stream>>>(
        bufC, inW, qkv, inB, NN, 3 * HID, HID, HID, HID, 3 * HID, 1.0f);

    // ---- attention: per-head scores -> softmax(+mean-A) -> A @ V ----
    k_fill<<<(int)(((size_t)NN * NN + EW - 1) / EW), EW, 0, stream>>>(
        meanA, 0.0f, NN * NN);
    const float qscale = 0.17677669529663688f;     // DH^-0.5
    for (int h = 0; h < HEADS; ++h) {
        const float* Qh = qkv + h * DH;                    // lda = 768
        const float* Kh = qkv + HID + h * DH;              // ldb = 768
        const float* Vh = qkv + 2 * HID + h * DH;          // ldb = 768
        // scores[N,N] = (qscale*Q_h) @ K_h^T
        gemm_wmma_f16<false, true><<<dim3(NN / 64, NN / 32), gb, 0, stream>>>(
            Qh, Kh, scores, nullptr, NN, NN, DH, 3 * HID, 3 * HID, NN, qscale);
        // softmax rows in place; meanA += A / HEADS
        k_softmax_meanacc<<<NN, EW, 0, stream>>>(scores, meanA);
        // attn[:, h*DH : (h+1)*DH] = A @ V_h
        gemm_wmma_f16<false, false><<<dim3(NN / 64, DH / 32), gb, 0, stream>>>(
            scores, Vh, attn + h * DH, nullptr, NN, DH, NN, NN, 3 * HID, HID, 1.0f);
    }

    // ---- output projection of attention ----
    gemm_wmma_f16<false, true><<<dim3(NN / 64, HID / 32), gb, 0, stream>>>(
        attn, outW, attn2, outB, NN, HID, HID, HID, HID, HID, 1.0f);

    // ---- attention-weighted pooling: pooled = meanA^T @ attn2 ----
    gemm_wmma_f16<true, false><<<dim3(NN / 64, HID / 32), gb, 0, stream>>>(
        meanA, attn2, pooled, nullptr, NN, HID, NN, NN, HID, HID, 1.0f);

    // ---- final projection: out[N,512] = pooled @ projW^T + projB ----
    gemm_wmma_f16<false, true><<<dim3(NN / 64, OUTD / 32), gb, 0, stream>>>(
        pooled, projW, out, projB, NN, OUTD, HID, HID, HID, OUTD, 1.0f);
}
